// DynamicDASBlock_31499290149183
// MI455X (gfx1250) — matrix operand order repaired and checked by reading, not compiled, not run on hardware
//
#include <hip/hip_runtime.h>
#include <hip/hip_bf16.h>

typedef __attribute__((ext_vector_type(16))) _Float16 v16h;
typedef __attribute__((ext_vector_type(8)))  _Float16 v8h;
typedef __attribute__((ext_vector_type(8)))  float    v8f;

#define Bn   16
#define Cn   256
#define HWn  16384   // 128*128
#define En   3
#define Dn   3
#define HIDn 16

// ---------------------------------------------------------------------------
// Kernel 1: routing MLP + dyn_w synthesis, output f16 [B][O][C] (C contiguous)
// grid = (64, 16): 64 chunks of 1024 elements per batch, blockIdx.y = batch
// ---------------------------------------------------------------------------
__global__ __launch_bounds__(256) void build_dynw_kernel(
    const float* __restrict__ scores,   // [B,3]
    const float* __restrict__ fc1_w,    // [3,16]
    const float* __restrict__ fc1_b,    // [16]
    const float* __restrict__ fc2_w,    // [16,3]
    const float* __restrict__ fc2_b,    // [3]
    const float* __restrict__ expert_w, // [3,256,256]
    _Float16* __restrict__ dynw)        // [B,256,256] f16
{
    __shared__ float rw[En];
    const int b = blockIdx.y;

    if (threadIdx.x == 0) {
        float h[HIDn];
        #pragma unroll
        for (int j = 0; j < HIDn; ++j) {
            float s = fc1_b[j];
            #pragma unroll
            for (int d = 0; d < Dn; ++d)
                s += scores[b * Dn + d] * fc1_w[d * HIDn + j];
            h[j] = s > 0.f ? s : 0.f;
        }
        float lg[En], mx = -1e30f;
        #pragma unroll
        for (int e = 0; e < En; ++e) {
            float s = fc2_b[e];
            #pragma unroll
            for (int j = 0; j < HIDn; ++j)
                s += h[j] * fc2_w[j * En + e];
            lg[e] = s;
            mx = s > mx ? s : mx;
        }
        float den = 0.f;
        #pragma unroll
        for (int e = 0; e < En; ++e) { lg[e] = __expf(lg[e] - mx); den += lg[e]; }
        const float inv = 1.f / den;
        #pragma unroll
        for (int e = 0; e < En; ++e) rw[e] = lg[e] * inv;
    }
    __syncthreads();

    const float r0 = rw[0], r1 = rw[1], r2 = rw[2];

    const int idx = (blockIdx.x * 256 + threadIdx.x) * 4;  // within [0, 65536)
    const float4 e0 = *(const float4*)(expert_w + 0 * Cn * Cn + idx);
    const float4 e1 = *(const float4*)(expert_w + 1 * Cn * Cn + idx);
    const float4 e2 = *(const float4*)(expert_w + 2 * Cn * Cn + idx);

    union { _Float16 h[4]; uint2 u; } pk;
    pk.h[0] = (_Float16)(r0 * e0.x + r1 * e1.x + r2 * e2.x);
    pk.h[1] = (_Float16)(r0 * e0.y + r1 * e1.y + r2 * e2.y);
    pk.h[2] = (_Float16)(r0 * e0.z + r1 * e1.z + r2 * e2.z);
    pk.h[3] = (_Float16)(r0 * e0.w + r1 * e1.w + r2 * e2.w);

    *(uint2*)(dynw + (size_t)b * Cn * Cn + idx) = pk.u;
}

// ---------------------------------------------------------------------------
// Kernel 2: out[b,o,hw] = x[b,o,hw] + sum_c dynw[b,o,c] * x[b,c,hw]
// One wave -> 16(o) x 16(hw) tile, K = 256 via 8x v_wmma_f32_16x16x32_f16.
// A (dyn_w rows) cached in 8 x v16h registers, reused over 8 hw-tiles/wave.
// grid = (16 hw-blocks, 16 o-tiles, 16 batches), block = 256 threads (8 waves)
// ---------------------------------------------------------------------------
__global__ __launch_bounds__(256) void moe_conv1x1_kernel(
    const float*    __restrict__ x,     // [B,256,HW]
    const _Float16* __restrict__ dynw,  // [B,256,256] f16
    float*          __restrict__ out)   // [B,256,HW]
{
    const int lane = threadIdx.x & 31;
    const int wave = threadIdx.x >> 5;
    const int half = lane >> 4;   // 0: lanes 0-15, 1: lanes 16-31
    const int l16  = lane & 15;

    const int b  = blockIdx.z;
    const int o0 = blockIdx.y * 16;

    const float*    xb   = x    + (size_t)b * Cn * HWn;
    float*          outb = out  + (size_t)b * Cn * HWn;
    const _Float16* wb   = dynw + (size_t)b * Cn * Cn;

    // ---- Load A tile (16 rows x K=256) into registers per ISA A-layout:
    // lane<16 : M=lane,  elems 0-7 = K kc*32+0..7,  elems 8-15 = K kc*32+16..23
    // lane>=16: M=l16,   elems 0-7 = K kc*32+8..15, elems 8-15 = K kc*32+24..31
    v16h A[8];
    const _Float16* wrow = wb + (size_t)(o0 + l16) * Cn;
    #pragma unroll
    for (int kc = 0; kc < 8; ++kc) {
        const v8h lo = *(const v8h*)(wrow + kc * 32 + half * 8);
        const v8h hi = *(const v8h*)(wrow + kc * 32 + half * 8 + 16);
        #pragma unroll
        for (int i = 0; i < 8; ++i) { A[kc][i] = lo[i]; A[kc][8 + i] = hi[i]; }
    }

    const int kbase = half * 8;   // B layout: lanes 0-15 hold K 0-7,16-23; 16-31 hold K 8-15,24-31

    #pragma unroll 1
    for (int t = 0; t < 8; ++t) {
        const int hw0 = (blockIdx.x * 64 + wave * 8 + t) * 16;
        v8f acc = {};

        #pragma unroll
        for (int kc = 0; kc < 8; ++kc) {
            const float* xcol = xb + (size_t)(kc * 32 + kbase) * HWn + hw0 + l16;
            v16h Bm;
            #pragma unroll
            for (int e = 0; e < 16; ++e)
                Bm[e] = (_Float16)xcol[(size_t)(e + (e >> 3) * 8) * HWn];

            acc = __builtin_amdgcn_wmma_f32_16x16x32_f16(
                      /*neg_a=*/false, A[kc], /*neg_b=*/false, Bm,
                      /*c_mod=*/(short)0, acc, /*reuse_a=*/false, /*reuse_b=*/false);
        }

        // D layout: VGPR r -> M = r + 8*half, N = l16. Residual add + store.
        #pragma unroll
        for (int r = 0; r < 8; ++r) {
            const int row   = o0 + r + half * 8;
            const size_t off = (size_t)row * HWn + hw0 + l16;
            outb[off] = acc[r] + xb[off];
        }
    }
}

// ---------------------------------------------------------------------------
extern "C" void kernel_launch(void* const* d_in, const int* in_sizes, int n_in,
                              void* d_out, int out_size, void* d_ws, size_t ws_size,
                              hipStream_t stream) {
    const float* x        = (const float*)d_in[0]; // [16,256,128,128]
    const float* scores   = (const float*)d_in[1]; // [16,3]
    const float* fc1_w    = (const float*)d_in[2]; // [3,16]
    const float* fc1_b    = (const float*)d_in[3]; // [16]
    const float* fc2_w    = (const float*)d_in[4]; // [16,3]
    const float* fc2_b    = (const float*)d_in[5]; // [3]
    const float* expert_w = (const float*)d_in[6]; // [3,256,256]

    _Float16* dynw = (_Float16*)d_ws;              // 16*256*256*2B = 2 MiB
    float*    out  = (float*)d_out;

    build_dynw_kernel<<<dim3(64, Bn), 256, 0, stream>>>(
        scores, fc1_w, fc1_b, fc2_w, fc2_b, expert_w, dynw);

    moe_conv1x1_kernel<<<dim3(16, 16, Bn), 256, 0, stream>>>(x, dynw, out);
}
